// GATEncoder_79757542686957
// MI455X (gfx1250) — matrix operand order repaired and checked by reading, not compiled, not run on hardware
//
#include <hip/hip_runtime.h>
#include <stdint.h>

typedef __attribute__((ext_vector_type(16))) __bf16 v16bf;
typedef __attribute__((ext_vector_type(8)))  __bf16 v8bf;
typedef __attribute__((ext_vector_type(8)))  float  v8f;

// ---------- helpers ----------
__device__ __forceinline__ int enc_f32(float f) {
  int i = __float_as_int(f);
  return (i >= 0) ? i : (i ^ 0x7fffffff);   // monotone float->int key
}
__device__ __forceinline__ float dec_f32(int i) {
  return __int_as_float((i >= 0) ? i : (i ^ 0x7fffffff));
}
__device__ __forceinline__ float lrelu(float v) { return v > 0.f ? v : 0.2f * v; }

// ---------- f32 -> bf16 conversion ----------
__global__ void k_f32_to_bf16(const float* __restrict__ in, __bf16* __restrict__ out, int n) {
  int i = blockIdx.x * blockDim.x + threadIdx.x;
  if (i < n) out[i] = (__bf16)in[i];
}

// ---------- zero-fill ----------
__global__ void k_zero(float* __restrict__ p, int n) {
  int i = blockIdx.x * blockDim.x + threadIdx.x;
  if (i < n) p[i] = 0.f;
}

// ---------- WMMA GEMM: C[M,Nc] = A[M,K] * Bt[Nc,K]^T  (bf16 in, f32 out) ----------
// Requirements: M%16==0, K%32==0, Nc%64==0. One wave -> 16x64 output strip.
// All 10 loads of a K-step are issued before the 4 WMMAs so VMEM latency
// overlaps the matrix pipe (partial s_wait_loadcnt instead of wait-to-zero).
__global__ __launch_bounds__(256) void k_wmma_gemm(
    const __bf16* __restrict__ A, const __bf16* __restrict__ Bt,
    float* __restrict__ C, int M, int Nc, int K) {
  const int lane = threadIdx.x & 31;
  const int wave = threadIdx.x >> 5;
  const int gw   = blockIdx.x * (blockDim.x >> 5) + wave;
  const int mtiles = M >> 4;
  const int ntiles = Nc >> 6;
  if (gw >= mtiles * ntiles) return;
  const int mt   = gw / ntiles;
  const int nt   = gw % ntiles;
  const int half = lane >> 4;       // 0: lanes 0-15, 1: lanes 16-31
  const int l16  = lane & 15;
  const int arow  = mt * 16 + l16;  // A row owned by this lane
  const int abase = half * 8;       // A: K base within 32-window (ISA 16-bit A layout)
  const int bbase = half * 8;       // B: K base within 32-window

  const __bf16* ap = A + (size_t)arow * K + abase;
  const __bf16* bp[4];
#pragma unroll
  for (int t = 0; t < 4; ++t) {
    const int bcol = nt * 64 + t * 16 + l16;
    bp[t] = Bt + (size_t)bcol * K + bbase;
  }

  v8f acc[4];
#pragma unroll
  for (int t = 0; t < 4; ++t) acc[t] = (v8f)(0.0f);

  for (int k0 = 0; k0 < K; k0 += 32) {
    // ---- issue all loads for this K-step first ----
    v8bf alo = *(const v8bf*)(ap);
    v8bf ahi = *(const v8bf*)(ap + 16);
    v8bf bl[4], bh[4];
#pragma unroll
    for (int t = 0; t < 4; ++t) {
      bl[t] = *(const v8bf*)(bp[t]);
      bh[t] = *(const v8bf*)(bp[t] + 16);
    }
    // prefetch next K-step of the streamed A operand (global_prefetch_b8)
    if (k0 + 32 < K) __builtin_prefetch(ap + 32, 0, 3);

    v16bf a;
#pragma unroll
    for (int i = 0; i < 8; ++i) { a[i] = alo[i]; a[i + 8] = ahi[i]; }

    // ---- 4 WMMAs back-to-back, reusing A ----
#pragma unroll
    for (int t = 0; t < 4; ++t) {
      v16bf b;
#pragma unroll
      for (int i = 0; i < 8; ++i) { b[i] = bl[t][i]; b[i + 8] = bh[t][i]; }
      acc[t] = __builtin_amdgcn_wmma_f32_16x16x32_bf16(
          /*neg_a=*/false, a, /*neg_b=*/false, b,
          /*c_mod=*/(short)0, acc[t], /*reuse_a=*/false, /*reuse_b=*/false);
    }

    ap += 32;
#pragma unroll
    for (int t = 0; t < 4; ++t) bp[t] += 32;
  }

  // C/D layout: VGPR r -> row mt*16 + half*8 + r, col nt*64 + t*16 + l16
  const int crow0 = mt * 16 + half * 8;
#pragma unroll
  for (int t = 0; t < 4; ++t) {
    const int col = nt * 64 + t * 16 + l16;
#pragma unroll
    for (int r = 0; r < 8; ++r) {
      C[(size_t)(crow0 + r) * Nc + col] = acc[t][r];
    }
  }
}

// ---------- attention coefficients + per-(node,head) init ----------
__global__ void k_attn(const float* __restrict__ h, const float* __restrict__ att_s,
                       const float* __restrict__ att_d,
                       float* __restrict__ a_s, float* __restrict__ a_d,
                       int* __restrict__ emax, float* __restrict__ denom,
                       int N, int H, int C) {
  int i = blockIdx.x * blockDim.x + threadIdx.x;  // over N*H
  if (i >= N * H) return;
  int n = i / H, hh = i % H;
  const float* hp  = h + (size_t)n * H * C + (size_t)hh * C;
  const float* asv = att_s + hh * C;
  const float* adv = att_d + hh * C;
  float ss = 0.f, dd = 0.f;
  for (int c = 0; c < C; ++c) { float v = hp[c]; ss += v * asv[c]; dd += v * adv[c]; }
  a_s[i] = ss;
  a_d[i] = dd;
  emax[i] = enc_f32(-3.0e38f);
  denom[i] = 0.f;
}

// ---------- edge pass 1: segment max of leaky-relu logits ----------
__global__ void k_edge_max(const int* __restrict__ ei, int E, int EL,
                           const float* __restrict__ a_s, const float* __restrict__ a_d,
                           int* __restrict__ emax, int H) {
  int i = blockIdx.x * blockDim.x + threadIdx.x;  // over EL*H
  if (i >= EL * H) return;
  int e = i / H, hh = i % H;
  int src = (e < E) ? ei[e] : (e - E);
  int dst = (e < E) ? ei[E + e] : (e - E);
  float v = lrelu(a_s[src * H + hh] + a_d[dst * H + hh]);
  atomicMax(&emax[dst * H + hh], enc_f32(v));
}

// ---------- edge pass 2: segment sum of exp(e - max) ----------
__global__ void k_edge_sum(const int* __restrict__ ei, int E, int EL,
                           const float* __restrict__ a_s, const float* __restrict__ a_d,
                           const int* __restrict__ emax, float* __restrict__ denom, int H) {
  int i = blockIdx.x * blockDim.x + threadIdx.x;  // over EL*H
  if (i >= EL * H) return;
  int e = i / H, hh = i % H;
  int src = (e < E) ? ei[e] : (e - E);
  int dst = (e < E) ? ei[E + e] : (e - E);
  float v = lrelu(a_s[src * H + hh] + a_d[dst * H + hh]);
  float m = dec_f32(emax[dst * H + hh]);
  atomicAdd(&denom[dst * H + hh], __expf(v - m));
}

// ---------- edge pass 3: alpha-weighted scatter-add of messages (wave per edge) ----------
__global__ __launch_bounds__(256) void k_edge_agg(
    const int* __restrict__ ei, int E, int EL,
    const float* __restrict__ a_s, const float* __restrict__ a_d,
    const int* __restrict__ emax, const float* __restrict__ denom,
    const float* __restrict__ h, float* __restrict__ agg, int H, int C) {
  int lane = threadIdx.x & 31;
  int wave = threadIdx.x >> 5;
  int e = blockIdx.x * (blockDim.x >> 5) + wave;
  if (e >= EL) return;
  int src = (e < E) ? ei[e] : (e - E);
  int dst = (e < E) ? ei[E + e] : (e - E);
  int HC = H * C;
  for (int c = lane; c < HC; c += 32) {
    int hh = c / C;
    float v = lrelu(a_s[src * H + hh] + a_d[dst * H + hh]);
    float m = dec_f32(emax[dst * H + hh]);
    float alpha = __expf(v - m) / (denom[dst * H + hh] + 1e-16f);
    atomicAdd(&agg[(size_t)dst * HC + c], h[(size_t)src * HC + c] * alpha);
  }
}

// ---------- layer-0 epilogue: bias + relu, emit bf16 for next GEMM ----------
__global__ void k_bias_relu_bf16(const float* __restrict__ agg, const float* __restrict__ b,
                                 __bf16* __restrict__ out, int n, int HC) {
  int i = blockIdx.x * blockDim.x + threadIdx.x;
  if (i < n) {
    float v = agg[i] + b[i % HC];
    out[i] = (__bf16)(v > 0.f ? v : 0.f);
  }
}

// ---------- layer-1 epilogue: bias add in place ----------
__global__ void k_bias_add(float* __restrict__ out, const float* __restrict__ b, int n, int C) {
  int i = blockIdx.x * blockDim.x + threadIdx.x;
  if (i < n) out[i] += b[i % C];
}

static inline int cdiv(int a, int b) { return (a + b - 1) / b; }

extern "C" void kernel_launch(void* const* d_in, const int* in_sizes, int n_in,
                              void* d_out, int out_size, void* d_ws, size_t ws_size,
                              hipStream_t stream) {
  (void)n_in; (void)out_size; (void)ws_size;
  const float* x   = (const float*)d_in[0];
  const int*   ei  = (const int*)  d_in[1];
  const float* W0  = (const float*)d_in[2];
  const float* as0 = (const float*)d_in[3];
  const float* ad0 = (const float*)d_in[4];
  const float* b0  = (const float*)d_in[5];
  const float* W1  = (const float*)d_in[6];
  const float* as1 = (const float*)d_in[7];
  const float* ad1 = (const float*)d_in[8];
  const float* b1  = (const float*)d_in[9];

  const int Fin = 128, H0 = 4, C0 = 64, HC0 = 256, C1 = 64;
  const int Nn = in_sizes[0] / Fin;     // 50000
  const int E  = in_sizes[1] / 2;       // 800000
  const int EL = E + Nn;                // + self loops

  // -------- workspace layout (256B aligned, with reuse) --------
  char* w = (char*)d_ws;
  size_t off = 0;
  auto alloc = [&](size_t bytes) -> void* {
    void* p = w + off;
    off += bytes;
    off = (off + 255) & ~(size_t)255;
    return p;
  };
  __bf16* xb   = (__bf16*)alloc((size_t)Nn * Fin * 2);   // bf16 x
  __bf16* W0b  = (__bf16*)alloc((size_t)HC0 * Fin * 2);  // bf16 W0
  __bf16* W1b  = (__bf16*)alloc((size_t)C1 * HC0 * 2);   // bf16 W1
  float*  h0   = (float*) alloc((size_t)Nn * HC0 * 4);   // layer0 features
  float*  vas0 = (float*) alloc((size_t)Nn * H0 * 4);
  float*  vad0 = (float*) alloc((size_t)Nn * H0 * 4);
  int*    em0  = (int*)   alloc((size_t)Nn * H0 * 4);
  float*  dn0  = (float*) alloc((size_t)Nn * H0 * 4);
  float*  agg0 = (float*) alloc((size_t)Nn * HC0 * 4);
  // reuse: bf16 relu output overlays h0 (h0 dead after layer-0 aggregation)
  __bf16* hrb  = (__bf16*)h0;
  // reuse: h1 overlays xb (same byte size; xb dead after GEMM0)
  float*  h1   = (float*)xb;
  // reuse: layer-1 scalars overlay layer-0 scalars (smaller)
  float*  vas1 = vas0;
  float*  vad1 = vad0;
  int*    em1  = em0;
  float*  dn1  = dn0;
  float*  out  = (float*)d_out;          // agg1 accumulates directly into output

  const int T = 256;

  // -------- precision conversion --------
  k_f32_to_bf16<<<cdiv(Nn * Fin, T), T, 0, stream>>>(x, xb, Nn * Fin);
  k_f32_to_bf16<<<cdiv(HC0 * Fin, T), T, 0, stream>>>(W0, W0b, HC0 * Fin);
  k_f32_to_bf16<<<cdiv(C1 * HC0, T), T, 0, stream>>>(W1, W1b, C1 * HC0);

  // -------- layer 0 --------
  {
    int jobs = (Nn / 16) * (HC0 / 64);
    k_wmma_gemm<<<cdiv(jobs, 8), 256, 0, stream>>>(xb, W0b, h0, Nn, HC0, Fin);
  }
  k_attn<<<cdiv(Nn * H0, T), T, 0, stream>>>(h0, as0, ad0, vas0, vad0, em0, dn0, Nn, H0, C0);
  k_zero<<<cdiv(Nn * HC0, T), T, 0, stream>>>(agg0, Nn * HC0);
  k_edge_max<<<cdiv(EL * H0, T), T, 0, stream>>>(ei, E, EL, vas0, vad0, em0, H0);
  k_edge_sum<<<cdiv(EL * H0, T), T, 0, stream>>>(ei, E, EL, vas0, vad0, em0, dn0, H0);
  k_edge_agg<<<cdiv(EL, 8), 256, 0, stream>>>(ei, E, EL, vas0, vad0, em0, dn0, h0, agg0, H0, C0);
  k_bias_relu_bf16<<<cdiv(Nn * HC0, T), T, 0, stream>>>(agg0, b0, hrb, Nn * HC0, HC0);

  // -------- layer 1 --------
  {
    int jobs = (Nn / 16) * (C1 / 64);
    k_wmma_gemm<<<cdiv(jobs, 8), 256, 0, stream>>>(hrb, W1b, h1, Nn, C1, HC0);
  }
  k_attn<<<cdiv(Nn, T), T, 0, stream>>>(h1, as1, ad1, vas1, vad1, em1, dn1, Nn, 1, C1);
  k_zero<<<cdiv(Nn * C1, T), T, 0, stream>>>(out, Nn * C1);
  k_edge_max<<<cdiv(EL, T), T, 0, stream>>>(ei, E, EL, vas1, vad1, em1, 1);
  k_edge_sum<<<cdiv(EL, T), T, 0, stream>>>(ei, E, EL, vas1, vad1, em1, dn1, 1);
  k_edge_agg<<<cdiv(EL, 8), 256, 0, stream>>>(ei, E, EL, vas1, vad1, em1, dn1, h1, out, 1, C1);
  k_bias_add<<<cdiv(Nn * C1, T), T, 0, stream>>>(out, b1, Nn * C1, C1);
}
